// PerResidualFP16ScaledTransformerBlock_72129680769366
// MI455X (gfx1250) — matrix operand order repaired and checked by reading, not compiled, not run on hardware
//
#include <hip/hip_runtime.h>
#include <hip/hip_bf16.h>

// ---------------------------------------------------------------------------
// Transformer block on MI455X (gfx1250, wave32, WMMA 16x16x32 f16).
// GEMMs: 256x128 workgroup tile, 8 waves x (64x64) -> 16 wmma / 16 ds_load_b128
// per K-step, double-buffered LDS with GLOBAL_LOAD_ASYNC_TO_LDS_B128 when the
// toolchain exposes it (ASYNCcnt pipeline), sync fallback otherwise.
// ---------------------------------------------------------------------------

typedef _Float16 v8h  __attribute__((ext_vector_type(8)));
typedef _Float16 v16h __attribute__((ext_vector_type(16)));
typedef float    v8f  __attribute__((ext_vector_type(8)));

#define B_   4
#define N_   2048
#define E_   1024
#define H_   16
#define D_   64
#define HID_ 4096
#define M_   (B_ * N_)   // 8192 rows through every GEMM

#ifndef __has_builtin
#define __has_builtin(x) 0
#endif

#if __has_builtin(__builtin_amdgcn_global_load_async_to_lds_b128)
#define HAVE_ASYNC 1
#else
#define HAVE_ASYNC 0
#endif

#if HAVE_ASYNC
// Builtin prototype (from hipcc diagnostic): first param is
//   int __attribute__((vector_size(16))) __device__ *   (global / AS1)
// second is the LDS destination (AS3), then imm offset, imm cpol.
typedef int v4i_gcc __attribute__((vector_size(16)));
typedef __attribute__((address_space(1))) v4i_gcc glb_v4i;
typedef __attribute__((address_space(3))) v4i_gcc lds_v4i;

// One 16-byte async global->LDS move per lane (tracked by ASYNCcnt).
__device__ __forceinline__ void async_cp16(const void* g, void* l) {
  __builtin_amdgcn_global_load_async_to_lds_b128(
      (glb_v4i*)(unsigned long long)g,
      (lds_v4i*)(unsigned int)(unsigned long long)l, 0, 0);
}
__device__ __forceinline__ void wait_async0() {
#if __has_builtin(__builtin_amdgcn_s_wait_asynccnt)
  __builtin_amdgcn_s_wait_asynccnt(0);
#else
  asm volatile("s_wait_asynccnt 0x0" ::: "memory");
#endif
}
#endif

__device__ __forceinline__ v8f zero8() {
  v8f z;
#pragma unroll
  for (int i = 0; i < 8; ++i) z[i] = 0.0f;
  return z;
}

__device__ __forceinline__ v16h combine16(v8h lo, v8h hi) {
  v16h r;
#pragma unroll
  for (int i = 0; i < 8; ++i) { r[i] = lo[i]; r[i + 8] = hi[i]; }
  return r;
}

// A-matrix 16x32 f16 fragment (ISA 7.12.2): lane (row = lane&15, half = lane>>4)
// holds K = [half*8, half*8+8) and [half*8+16, half*8+24): two 16B chunks.
__device__ __forceinline__ v16h load_fragA(const _Float16* tile, int ld, int mo, int lane) {
  const int row = lane & 15, half = lane >> 4;
  const _Float16* p = tile + (size_t)(mo + row) * ld + half * 8;
  return combine16(*(const v8h*)p, *(const v8h*)(p + 16));
}

// B-matrix 32x16 f16 fragment: lane (col = lane&15, half = lane>>4) holds
// K = [half*8, half*8+8) and [half*8+16, half*8+24): two 16B chunks of the column.
__device__ __forceinline__ v16h load_fragB(const _Float16* tile, int ld, int no, int lane) {
  const int col = lane & 15, half = lane >> 4;
  const _Float16* p = tile + (size_t)(no + col) * ld + half * 8;
  return combine16(*(const v8h*)p, *(const v8h*)(p + 16));
}

// ---------------------------------------------------------------------------
// fp32 [K,N] row-major  ->  fp16 [N,K] row-major (column-major B for WMMA)
// ---------------------------------------------------------------------------
__global__ __launch_bounds__(256)
void transpose_f32_to_f16(const float* __restrict__ W, _Float16* __restrict__ Bt,
                          int K, int Nn) {
  __shared__ float tile[32][33];
  const int n0 = blockIdx.x * 32, k0 = blockIdx.y * 32;
  const int tx = threadIdx.x & 31, ty = threadIdx.x >> 5;  // 32 x 8
#pragma unroll
  for (int j = 0; j < 32; j += 8)
    tile[ty + j][tx] = W[(size_t)(k0 + ty + j) * Nn + n0 + tx];
  __syncthreads();
#pragma unroll
  for (int j = 0; j < 32; j += 8)
    Bt[(size_t)(n0 + ty + j) * K + k0 + tx] = (_Float16)tile[tx][ty + j];
}

// ---------------------------------------------------------------------------
// LayerNorm (fp32 in) -> fp16 out. One block per row of E=1024.
// ---------------------------------------------------------------------------
__global__ __launch_bounds__(256)
void layernorm_f16(const float* __restrict__ x, const float* __restrict__ g,
                   const float* __restrict__ b, _Float16* __restrict__ out) {
  const int row = blockIdx.x;
  const float* xr = x + (size_t)row * E_;
  float s = 0.0f, ss = 0.0f;
  for (int i = threadIdx.x; i < E_; i += 256) {
    float v = xr[i];
    s += v; ss += v * v;
  }
#pragma unroll
  for (int o = 16; o > 0; o >>= 1) {
    s  += __shfl_xor(s,  o, 32);
    ss += __shfl_xor(ss, o, 32);
  }
  __shared__ float ws[8], wss[8];
  const int wave = threadIdx.x >> 5, lane = threadIdx.x & 31;
  if (lane == 0) { ws[wave] = s; wss[wave] = ss; }
  __syncthreads();
  if (wave == 0) {
    s  = (lane < 8) ? ws[lane]  : 0.0f;
    ss = (lane < 8) ? wss[lane] : 0.0f;
#pragma unroll
    for (int o = 4; o > 0; o >>= 1) {
      s  += __shfl_xor(s,  o, 32);
      ss += __shfl_xor(ss, o, 32);
    }
    if (lane == 0) { ws[0] = s; wss[0] = ss; }
  }
  __syncthreads();
  const float mean = ws[0] * (1.0f / E_);
  const float var  = wss[0] * (1.0f / E_) - mean * mean;
  const float rstd = rsqrtf(var + 1e-5f);
  _Float16* orow = out + (size_t)row * E_;
  for (int i = threadIdx.x; i < E_; i += 256)
    orow[i] = (_Float16)((xr[i] - mean) * rstd * g[i] + b[i]);
}

// ---------------------------------------------------------------------------
// WMMA GEMM: C[M,N] = A[M,K](f16,row) x B (f16, Bt[N,K]) + bias, epilogue.
// 256 threads = 8 waves. Workgroup tile 256(M)x128(N); wave tile 64x64.
// K stepped 32 at a time, LDS double-buffered.
// ---------------------------------------------------------------------------
enum { EPI_F16 = 0, EPI_GELU = 1, EPI_RES = 2 };

template <int EPI>
__global__ __launch_bounds__(256)
void gemm_wmma(const _Float16* __restrict__ A, const _Float16* __restrict__ Bt,
               const float* __restrict__ bias, const float* __restrict__ resid,
               void* __restrict__ Cout, int Mdim, int Ndim, int Kdim) {
  constexpr int LD = 40;  // 32 + 8 halves pad (80B rows, 16B aligned)
  __shared__ __align__(16) _Float16 As[2][256 * LD];   // 2 x 20 KB
  __shared__ __align__(16) _Float16 Bs[2][128 * LD];   // 2 x 10 KB

  const int t = threadIdx.x;
  const int lane = t & 31, wave = t >> 5;
  const int wm = (wave >> 1) * 64;          // wave M offset: 0/64/128/192
  const int wn = (wave & 1) * 64;           // wave N offset: 0/64
  const int m0 = blockIdx.y * 256;
  const int n0 = blockIdx.x * 128;

  v8f acc[4][4];
#pragma unroll
  for (int i = 0; i < 4; ++i)
#pragma unroll
    for (int j = 0; j < 4; ++j) acc[i][j] = zero8();

  // Cooperative staging indices: A thread owns a full 32-half row (64B),
  // B thread owns half a row (32B).
  const int arow = t;                 // 0..255
  const int brow = t >> 1;            // 0..127
  const int bseg = (t & 1) * 16;      // 0 / 16 halves

#if HAVE_ASYNC
  // ---- async double-buffered pipeline ----
  {
    const _Float16* ga = A  + (size_t)(m0 + arow) * Kdim;
    const _Float16* gb = Bt + (size_t)(n0 + brow) * Kdim + bseg;
#pragma unroll
    for (int s = 0; s < 4; ++s)
      async_cp16(ga + s * 8, &As[0][arow * LD + s * 8]);
#pragma unroll
    for (int s = 0; s < 2; ++s)
      async_cp16(gb + s * 8, &Bs[0][brow * LD + bseg + s * 8]);
  }
  int buf = 0;
  for (int kb = 0; kb < Kdim; kb += 32) {
    wait_async0();
    __syncthreads();
    if (kb + 32 < Kdim) {
      const _Float16* ga = A  + (size_t)(m0 + arow) * Kdim + kb + 32;
      const _Float16* gb = Bt + (size_t)(n0 + brow) * Kdim + kb + 32 + bseg;
#pragma unroll
      for (int s = 0; s < 4; ++s)
        async_cp16(ga + s * 8, &As[buf ^ 1][arow * LD + s * 8]);
#pragma unroll
      for (int s = 0; s < 2; ++s)
        async_cp16(gb + s * 8, &Bs[buf ^ 1][brow * LD + bseg + s * 8]);
      __builtin_prefetch(ga + 64, 0, 1);  // 2 k-tiles ahead -> global_prefetch_b8
    }
    v16h bf[4];
#pragma unroll
    for (int j = 0; j < 4; ++j) bf[j] = load_fragB(&Bs[buf][0], LD, wn + j * 16, lane);
#pragma unroll
    for (int i = 0; i < 4; ++i) {
      v16h af = load_fragA(&As[buf][0], LD, wm + i * 16, lane);
#pragma unroll
      for (int j = 0; j < 4; ++j)
        acc[i][j] = __builtin_amdgcn_wmma_f32_16x16x32_f16(false, af, false, bf[j],
                                                           (short)0, acc[i][j], false, false);
    }
    __syncthreads();
    buf ^= 1;
  }
#else
  // ---- synchronous fallback ----
  for (int kb = 0; kb < Kdim; kb += 32) {
    const _Float16* ga = A  + (size_t)(m0 + arow) * Kdim + kb;
    const _Float16* gb = Bt + (size_t)(n0 + brow) * Kdim + kb + bseg;
    __builtin_prefetch(ga + 64, 0, 1);
#pragma unroll
    for (int s = 0; s < 4; ++s)
      *(v8h*)(&As[0][arow * LD + s * 8]) = *(const v8h*)(ga + s * 8);
#pragma unroll
    for (int s = 0; s < 2; ++s)
      *(v8h*)(&Bs[0][brow * LD + bseg + s * 8]) = *(const v8h*)(gb + s * 8);
    __syncthreads();
    v16h bf[4];
#pragma unroll
    for (int j = 0; j < 4; ++j) bf[j] = load_fragB(&Bs[0][0], LD, wn + j * 16, lane);
#pragma unroll
    for (int i = 0; i < 4; ++i) {
      v16h af = load_fragA(&As[0][0], LD, wm + i * 16, lane);
#pragma unroll
      for (int j = 0; j < 4; ++j)
        acc[i][j] = __builtin_amdgcn_wmma_f32_16x16x32_f16(false, af, false, bf[j],
                                                           (short)0, acc[i][j], false, false);
    }
    __syncthreads();
  }
#endif

  // Epilogue. C/D layout: lane holds col n = lane&15, rows m = (lane>>4)*8 + r.
  const int cn = lane & 15, ch = lane >> 4;
#pragma unroll
  for (int i = 0; i < 4; ++i) {
#pragma unroll
    for (int j = 0; j < 4; ++j) {
      const int gm = m0 + wm + i * 16 + ch * 8;
      const int gn = n0 + wn + j * 16 + cn;
      const float bv = bias[gn];
#pragma unroll
      for (int r = 0; r < 8; ++r) {
        const float v = acc[i][j][r] + bv;
        const size_t idx = (size_t)(gm + r) * Ndim + gn;
        if (EPI == EPI_F16) {
          ((_Float16*)Cout)[idx] = (_Float16)v;
        } else if (EPI == EPI_GELU) {
          ((_Float16*)Cout)[idx] = (_Float16)(0.5f * v * (1.0f + erff(v * 0.70710678f)));
        } else {
          ((float*)Cout)[idx] = v + resid[idx];
        }
      }
    }
  }
}

// ---------------------------------------------------------------------------
// Flash-style attention. qkv: [B, N, 3E] f16. One block = (b,h) x 64 query rows.
// 128 threads = 4 waves; each wave owns 16 query rows, full D=64.
// 64-key blocks: 8 wmma (scores) + 8 wmma (PV) per softmax pass.
// No 1/sqrt(D) scale (reference omits it).
// ---------------------------------------------------------------------------
__global__ __launch_bounds__(128)
void attn_flash(const _Float16* __restrict__ qkv, _Float16* __restrict__ O) {
  constexpr int LDK = 72;  // 64 + 8 halves (144B rows)
  constexpr int LDV = 72;  // key-block 64 + 8
  constexpr int LDP = 72;
  __shared__ __align__(16) _Float16 Ks[64 * LDK];       // [key][d]
  __shared__ __align__(16) _Float16 Vt[64 * LDV];       // [d][key] (V transposed)
  __shared__ __align__(16) _Float16 Ps[4][16 * LDP];    // per-wave P [row][key]

  const int bh = blockIdx.y;
  const int b = bh / H_, h = bh % H_;
  const int qbase = blockIdx.x * 64;
  const int t = threadIdx.x, lane = t & 31, wave = t >> 5;
  const int frow = lane & 15, fhalf = lane >> 4;

  const size_t rowstride = 3 * E_;
  const _Float16* qptr = qkv + (size_t)b * N_ * rowstride + h * D_;
  const _Float16* kptr = qptr + E_;
  const _Float16* vptr = qptr + 2 * E_;

  // Q fragments, reused across all key blocks. A-layout, 2 k-chunks of 32.
  v16h qf[2];
  {
    const _Float16* qp = qptr + (size_t)(qbase + wave * 16 + frow) * rowstride;
#pragma unroll
    for (int kc = 0; kc < 2; ++kc) {
      const _Float16* p = qp + kc * 32 + fhalf * 8;
      qf[kc] = combine16(*(const v8h*)p, *(const v8h*)(p + 16));
    }
  }

  float mrow[8], lsum[8];
  v8f oacc[4];
#pragma unroll
  for (int r = 0; r < 8; ++r) { mrow[r] = -1e30f; lsum[r] = 0.0f; }
#pragma unroll
  for (int nt = 0; nt < 4; ++nt) oacc[nt] = zero8();

  for (int j0 = 0; j0 < N_; j0 += 64) {
    // Stage K block [64 keys][64 d] row-major and V block transposed [d][key].
    {
      const int key = t >> 1;            // 0..63
      const int seg = (t & 1) * 32;      // 0 / 32 halves
      const _Float16* kp = kptr + (size_t)(j0 + key) * rowstride + seg;
#pragma unroll
      for (int s = 0; s < 4; ++s)
        *(v8h*)(Ks + key * LDK + seg + s * 8) = *(const v8h*)(kp + s * 8);
      const _Float16* vp = vptr + (size_t)(j0 + key) * rowstride + seg;
#pragma unroll
      for (int s = 0; s < 4; ++s) {
        v8h vv = *(const v8h*)(vp + s * 8);
#pragma unroll
        for (int i = 0; i < 8; ++i)
          Vt[(seg + s * 8 + i) * LDV + key] = vv[i];
      }
    }
    __syncthreads();

    // Scores S[16 x 64] = Q[16 x 64] x K^T (B-frag columns = contiguous K rows).
    v8f s[4];
#pragma unroll
    for (int jt = 0; jt < 4; ++jt) s[jt] = zero8();
#pragma unroll
    for (int jt = 0; jt < 4; ++jt) {
#pragma unroll
      for (int kc = 0; kc < 2; ++kc) {
        const _Float16* p = Ks + (jt * 16 + frow) * LDK + kc * 32 + fhalf * 8;
        v16h bf = combine16(*(const v8h*)p, *(const v8h*)(p + 16));
        s[jt] = __builtin_amdgcn_wmma_f32_16x16x32_f16(false, qf[kc], false, bf,
                                                       (short)0, s[jt], false, false);
      }
    }

    // Online softmax. Row m = fhalf*8 + r lives across the 16 lanes of this half.
    _Float16* myP = &Ps[wave][0];
    float alpha[8];
#pragma unroll
    for (int r = 0; r < 8; ++r) {
      float mx = fmaxf(fmaxf(s[0][r], s[1][r]), fmaxf(s[2][r], s[3][r]));
#pragma unroll
      for (int o = 1; o < 16; o <<= 1) mx = fmaxf(mx, __shfl_xor(mx, o, 32));
      const float mnew = fmaxf(mrow[r], mx);
      alpha[r] = __expf(mrow[r] - mnew);
      mrow[r] = mnew;
      const int prow = fhalf * 8 + r;
      float ps = 0.0f;
#pragma unroll
      for (int jt = 0; jt < 4; ++jt) {
        const float pv = __expf(s[jt][r] - mnew);
        myP[prow * LDP + jt * 16 + frow] = (_Float16)pv;
        ps += pv;
      }
#pragma unroll
      for (int o = 1; o < 16; o <<= 1) ps += __shfl_xor(ps, o, 32);
      lsum[r] = lsum[r] * alpha[r] + ps;
    }
#pragma unroll
    for (int nt = 0; nt < 4; ++nt)
#pragma unroll
      for (int r = 0; r < 8; ++r) oacc[nt][r] *= alpha[r];

    // O += P x V. P reloaded from LDS in A-layout (per-wave region, DS in-order).
    v16h pf[2];
#pragma unroll
    for (int kc = 0; kc < 2; ++kc) {
      const _Float16* pp = myP + frow * LDP + kc * 32 + fhalf * 8;
      pf[kc] = combine16(*(const v8h*)pp, *(const v8h*)(pp + 16));
    }
#pragma unroll
    for (int nt = 0; nt < 4; ++nt) {
#pragma unroll
      for (int kc = 0; kc < 2; ++kc) {
        const _Float16* vv = Vt + (nt * 16 + frow) * LDV + kc * 32 + fhalf * 8;
        v16h bf = combine16(*(const v8h*)vv, *(const v8h*)(vv + 16));
        oacc[nt] = __builtin_amdgcn_wmma_f32_16x16x32_f16(false, pf[kc], false, bf,
                                                          (short)0, oacc[nt], false, false);
      }
    }
    __syncthreads();
  }

  // Normalize and store fp16 output at [b, n, h*D + d].
#pragma unroll
  for (int nt = 0; nt < 4; ++nt) {
#pragma unroll
    for (int r = 0; r < 8; ++r) {
      const int m = qbase + wave * 16 + fhalf * 8 + r;
      const int d = nt * 16 + frow;
      O[((size_t)b * N_ + m) * E_ + h * D_ + d] = (_Float16)(oacc[nt][r] / lsum[r]);
    }
  }
}

// ---------------------------------------------------------------------------
// Host-side orchestration (graph-capture safe: launches only).
// ---------------------------------------------------------------------------
extern "C" void kernel_launch(void* const* d_in, const int* in_sizes, int n_in,
                              void* d_out, int out_size, void* d_ws, size_t ws_size,
                              hipStream_t stream) {
  (void)in_sizes; (void)n_in; (void)out_size; (void)ws_size;

  const float* x     = (const float*)d_in[0];
  const float* ln1_g = (const float*)d_in[1];
  const float* ln1_b = (const float*)d_in[2];
  const float* wqkv  = (const float*)d_in[3];
  const float* bqkv  = (const float*)d_in[4];
  const float* wproj = (const float*)d_in[5];
  const float* bproj = (const float*)d_in[6];
  const float* ln2_g = (const float*)d_in[7];
  const float* ln2_b = (const float*)d_in[8];
  const float* w0    = (const float*)d_in[9];
  const float* b0    = (const float*)d_in[10];
  const float* w1    = (const float*)d_in[11];
  const float* b1    = (const float*)d_in[12];
  float* out = (float*)d_out;

  char* ws = (char*)d_ws;
  size_t off = 0;
  auto alloc = [&](size_t bytes) -> void* {
    void* p = ws + off;
    off += (bytes + 255) & ~(size_t)255;
    return p;
  };
  _Float16* h1     = (_Float16*)alloc((size_t)M_ * E_ * 2);
  _Float16* wqkvT  = (_Float16*)alloc((size_t)E_ * 3 * E_ * 2);
  _Float16* wprojT = (_Float16*)alloc((size_t)E_ * E_ * 2);
  _Float16* w0T    = (_Float16*)alloc((size_t)E_ * HID_ * 2);
  _Float16* w1T    = (_Float16*)alloc((size_t)HID_ * E_ * 2);
  _Float16* qkvb   = (_Float16*)alloc((size_t)M_ * 3 * E_ * 2);
  _Float16* aout   = (_Float16*)alloc((size_t)M_ * E_ * 2);
  float*    out1   = (float*)   alloc((size_t)M_ * E_ * 4);
  _Float16* h2     = (_Float16*)alloc((size_t)M_ * E_ * 2);
  _Float16* z      = (_Float16*)alloc((size_t)M_ * HID_ * 2);

  // Weight convert+transpose to fp16 column-major (WMMA B operand layout).
  transpose_f32_to_f16<<<dim3(3 * E_ / 32, E_ / 32), 256, 0, stream>>>(wqkv, wqkvT, E_, 3 * E_);
  transpose_f32_to_f16<<<dim3(E_ / 32,     E_ / 32), 256, 0, stream>>>(wproj, wprojT, E_, E_);
  transpose_f32_to_f16<<<dim3(HID_ / 32,   E_ / 32), 256, 0, stream>>>(w0, w0T, E_, HID_);
  transpose_f32_to_f16<<<dim3(E_ / 32, HID_ / 32), 256, 0, stream>>>(w1, w1T, HID_, E_);

  // h1 = LN1(x) -> fp16
  layernorm_f16<<<M_, 256, 0, stream>>>(x, ln1_g, ln1_b, h1);
  // qkv = h1 @ Wqkv + bqkv (f16 out)
  gemm_wmma<EPI_F16><<<dim3(3 * E_ / 128, M_ / 256), 256, 0, stream>>>(
      h1, wqkvT, bqkv, nullptr, qkvb, M_, 3 * E_, E_);
  // attention (flash, fp16 wmma)
  attn_flash<<<dim3(N_ / 64, B_ * H_), 128, 0, stream>>>(qkvb, aout);
  // out1 = attn @ Wproj + bproj + x (f32 out, residual)
  gemm_wmma<EPI_RES><<<dim3(E_ / 128, M_ / 256), 256, 0, stream>>>(
      aout, wprojT, bproj, x, out1, M_, E_, E_);
  // h2 = LN2(out1) -> fp16
  layernorm_f16<<<M_, 256, 0, stream>>>(out1, ln2_g, ln2_b, h2);
  // z = gelu(h2 @ W0 + b0) (f16 out)
  gemm_wmma<EPI_GELU><<<dim3(HID_ / 128, M_ / 256), 256, 0, stream>>>(
      h2, w0T, b0, nullptr, z, M_, HID_, E_);
  // out = z @ W1 + b1 + out1 (f32 out, residual)
  gemm_wmma<EPI_RES><<<dim3(E_ / 128, M_ / 256), 256, 0, stream>>>(
      z, w1T, b1, out1, out, M_, E_, HID_);
}
